// VectorQuantizer_69913477644917
// MI455X (gfx1250) — matrix operand order repaired and checked by reading, not compiled, not run on hardware
//
#include <hip/hip_runtime.h>

// ---------------- problem constants ----------------
#define BB    16
#define CC    256
#define TT    1500
#define KK    2048
#define NN    (BB * TT)          // 24000 query rows
#define QE    (NN * CC)          // 6,144,000 quantized elements
#define MT    (NN / 16)          // 1500 M-tiles (exact)
#define KSTEP 8                  // 256 / 32

// GEMM LDS staging: 64 codebook rows (4 N-tiles) per stage, double buffered
#define STAGE_ROWS 64
#define NSTAGES    (KK / STAGE_ROWS)   // 32
#define LDS_ROW_H  264                 // 256 halfs + 8 pad -> bank-conflict-free
#define STAGE_H    (STAGE_ROWS * LDS_ROW_H)

typedef __attribute__((ext_vector_type(16))) _Float16 v16h;
typedef __attribute__((ext_vector_type(8)))  _Float16 v8h;
typedef __attribute__((ext_vector_type(8)))  float    v8f;

union V16 { v16h v; v8h h[2]; };

// ---------------- workspace layout (bytes) ----------------
#define WS_XT    0                         // 24000*256 f16 = 12,288,000 B
#define WS_CB    12288000                  // 2048*256 f16  =  1,048,576 B
#define WS_EN    13336576                  // 2048 f32      =      8,192 B
#define WS_IDX   13344768                  // 24000 i32     =     96,000 B
#define WS_PART  13440768                  // 24000 f32     =     96,000 B

// ============ 1. transpose (B,C,T) f32 -> (B*T, C) f16 =================
__global__ void vq_transpose_x(const float* __restrict__ x,
                               _Float16* __restrict__ xt) {
  __shared__ float tile[32][33];
  int t0 = blockIdx.x * 32;
  int c0 = blockIdx.y * 32;
  int b  = blockIdx.z;
  int tx = threadIdx.x, ty = threadIdx.y;

  int t = t0 + tx, c = c0 + ty;
  if (t < TT)
    tile[ty][tx] = x[((size_t)b * CC + c) * TT + t];
  __syncthreads();

  int tw = t0 + ty;               // row index uses ty
  int cw = c0 + tx;               // column (coalesced)
  if (tw < TT)
    xt[((size_t)b * TT + tw) * CC + cw] = (_Float16)tile[tx][ty];
}

// ============ 2. codebook f32 -> f16 + ||e||^2 ========================
__global__ void vq_cb_prep(const float* __restrict__ cb,
                           _Float16* __restrict__ cbh,
                           float* __restrict__ enorm) {
  __shared__ float red[256];
  int k = blockIdx.x;
  int c = threadIdx.x;
  float v = cb[(size_t)k * CC + c];
  cbh[(size_t)k * CC + c] = (_Float16)v;
  red[c] = v * v;
  __syncthreads();
  for (int s = 128; s > 0; s >>= 1) {
    if (c < s) red[c] += red[c + s];
    __syncthreads();
  }
  if (c == 0) enorm[k] = red[0];
}

// ============ 3. WMMA GEMM + fused argmin =============================
// One wave per 16-row M-tile. Codebook B-tiles staged into LDS once per
// block via async global->LDS (double buffered, ASYNCcnt), shared by all
// 8 waves. B fragments double-buffered in registers so DS latency
// overlaps WMMA execution. 8 x v_wmma_f32_16x16x32_f16 per N-tile.
__global__ __launch_bounds__(256) void vq_wmma_argmin(
    const _Float16* __restrict__ xt,     // [24000][256]
    const _Float16* __restrict__ cbh,    // [2048][256]
    const float*    __restrict__ enorm,  // [2048]
    int*            __restrict__ idx_ws, // [24000]
    float*          __restrict__ idx_f)  // d_out + QE
{
  __shared__ _Float16 bsm[2 * STAGE_H];  // 2 x 33,792 B

  int wave = threadIdx.x >> 5;
  int lane = threadIdx.x & 31;
  int tile = blockIdx.x * 8 + wave;
  bool active = (tile < MT);
  int ctile = active ? tile : (MT - 1);  // clamp: all waves hit barriers

  int half = lane >> 4;                  // 0: lanes 0-15, 1: lanes 16-31
  int l15  = lane & 31 & 15;

  unsigned bsm_base = (unsigned)(unsigned long long)(void*)&bsm[0];

  // ---- async copy of one 64-row codebook stage into LDS buffer ----
  auto issue_stage = [&](int stage, int bufsel) {
    const _Float16* src = cbh + (size_t)stage * STAGE_ROWS * CC;
    unsigned dbase = bsm_base + (unsigned)(bufsel * STAGE_H * 2);
    // 2048 x 16B chunks per stage, 8 per thread
#pragma unroll
    for (int jj = 0; jj < 8; ++jj) {
      int q   = threadIdx.x + jj * 256;
      int row = q >> 5;                  // 0..63
      int col = q & 31;                  // 16B chunk within row
      unsigned dst = dbase + (unsigned)(row * (LDS_ROW_H * 2) + col * 16);
      unsigned long long g =
          (unsigned long long)(const void*)(src + row * CC + col * 8);
      asm volatile("global_load_async_to_lds_b128 %0, %1, off"
                   :: "v"(dst), "v"(g) : "memory");
    }
  };

  // ---- preload A tile: 16 rows x 256 cols, WMMA 16-bit A layout ----
  // lane(l15,half) holds row (ctile*16+l15); per k-step s (32 K):
  //   elems 0-7  = K s*32 + half*8 .. +7
  //   elems 8-15 = K s*32 + 16 + half*8 .. +7
  const _Float16* xrow = xt + (size_t)(ctile * 16 + l15) * CC + half * 8;
  V16 A[KSTEP];
#pragma unroll
  for (int s = 0; s < KSTEP; ++s) {
    A[s].h[0] = *(const v8h*)(xrow + s * 32);
    A[s].h[1] = *(const v8h*)(xrow + s * 32 + 16);
  }

  float minv[8];
  int   mini[8];
#pragma unroll
  for (int j = 0; j < 8; ++j) { minv[j] = 3.0e38f; mini[j] = 0; }

  // ---- prologue: stage 0 ----
  issue_stage(0, 0);
  asm volatile("s_wait_asynccnt 0x0" ::: "memory");
  __syncthreads();

  // ---- sweep codebook: 32 stages x 4 N-tiles ----
  for (int stage = 0; stage < NSTAGES; ++stage) {
    int bufsel = stage & 1;
    if (stage + 1 < NSTAGES) issue_stage(stage + 1, bufsel ^ 1);

    const _Float16* bbuf = bsm + bufsel * STAGE_H;

    // hoist ||e||^2 loads: L2 latency overlaps the first WMMAs
    float en[4];
#pragma unroll
    for (int u = 0; u < 4; ++u)
      en[u] = enorm[stage * STAGE_ROWS + u * 16 + l15];

#pragma unroll
    for (int u = 0; u < 4; ++u) {
      int nidx = stage * STAGE_ROWS + u * 16 + l15;  // lane's codebook row
      // B layout (32x16 f16): lane holds column l15, K s*32 + half*8 .. +7
      // and s*32 + 16 + half*8 .. +7 (two 16B LDS reads per k-step).
      const _Float16* brow = bbuf + (u * 16 + l15) * LDS_ROW_H + half * 8;

      // register double-buffer: load B[s+1] before WMMA on B[s] so the
      // compiler can wait dscnt<=2 and overlap DS latency with WMMA.
      V16 Bb[2];
      Bb[0].h[0] = *(const v8h*)(brow);
      Bb[0].h[1] = *(const v8h*)(brow + 16);

      v8f acc = {};
#pragma unroll
      for (int s = 0; s < KSTEP; ++s) {
        if (s + 1 < KSTEP) {
          Bb[(s + 1) & 1].h[0] = *(const v8h*)(brow + (s + 1) * 32);
          Bb[(s + 1) & 1].h[1] = *(const v8h*)(brow + (s + 1) * 32 + 16);
        }
        acc = __builtin_amdgcn_wmma_f32_16x16x32_f16(
            false, A[s].v, false, Bb[s & 1].v, (short)0, acc, false, false);
      }

      // C layout: VGPR j -> M = j + 8*half, N = l15.
      // distance (minus const ||x||^2):  d = ||e||^2 - 2 * x.e
#pragma unroll
      for (int j = 0; j < 8; ++j) {
        float d = __builtin_fmaf(-2.0f, acc[j], en[u]);
        if (d < minv[j]) { minv[j] = d; mini[j] = nidx; }
      }
    }

    // copies for stage+1 done, all waves done reading old buffer
    asm volatile("s_wait_asynccnt 0x0" ::: "memory");
    __syncthreads();
  }

  // ---- cross-lane argmin within each 16-lane half (prefer lower index) ----
#pragma unroll
  for (int j = 0; j < 8; ++j) {
    float v = minv[j];
    int   i = mini[j];
#pragma unroll
    for (int m = 1; m < 16; m <<= 1) {
      float ov = __shfl_xor(v, m, 16);
      int   oi = __shfl_xor(i, m, 16);
      if (ov < v || (ov == v && oi < i)) { v = ov; i = oi; }
    }
    if (active && l15 == 0) {
      int r = tile * 16 + j + 8 * half;  // rows 0-7 lane0, 8-15 lane16
      idx_ws[r] = i;
      idx_f[r]  = (float)i;
    }
  }
}

// ============ 4. gather quantized + loss partials =====================
__global__ void vq_gather(const float* __restrict__ x,
                          const float* __restrict__ cb,
                          const int*   __restrict__ idx_ws,
                          float* __restrict__ q_out,
                          float* __restrict__ partials) {
  __shared__ float red[256];
  int i = blockIdx.x * 256 + threadIdx.x;   // 24000 blocks * 256 == QE exactly
  int b   = i / (CC * TT);
  int rem = i - b * (CC * TT);
  int c   = rem / TT;
  int t   = rem - c * TT;
  int idx = idx_ws[b * TT + t];
  float q = cb[(size_t)idx * CC + c];
  q_out[i] = q;                              // quantized_out == codebook[idx]
  float diff = q - x[i];
  red[threadIdx.x] = diff * diff;
  __syncthreads();
  for (int s = 128; s > 0; s >>= 1) {
    if (threadIdx.x < s) red[threadIdx.x] += red[threadIdx.x + s];
    __syncthreads();
  }
  if (threadIdx.x == 0) partials[blockIdx.x] = red[0];
}

// ============ 5. deterministic loss finalize ==========================
__global__ void vq_loss_final(const float* __restrict__ partials,
                              float* __restrict__ loss_out) {
  __shared__ float red[256];
  float s = 0.0f;
  for (int i = threadIdx.x; i < NN; i += 256) s += partials[i];
  red[threadIdx.x] = s;
  __syncthreads();
  for (int k = 128; k > 0; k >>= 1) {
    if (threadIdx.x < k) red[threadIdx.x] += red[threadIdx.x + k];
    __syncthreads();
  }
  if (threadIdx.x == 0) loss_out[0] = red[0] * (1.0f / (float)QE);
}

// ======================================================================
extern "C" void kernel_launch(void* const* d_in, const int* in_sizes, int n_in,
                              void* d_out, int out_size, void* d_ws, size_t ws_size,
                              hipStream_t stream) {
  const float* x  = (const float*)d_in[0];   // (16,256,1500) f32
  const float* cb = (const float*)d_in[1];   // (2048,256) f32
  float* out = (float*)d_out;                // [QE quant][NN idx][1 loss]

  char* ws = (char*)d_ws;
  _Float16* xt     = (_Float16*)(ws + WS_XT);
  _Float16* cbh    = (_Float16*)(ws + WS_CB);
  float*    enorm  = (float*)   (ws + WS_EN);
  int*      idx_ws = (int*)     (ws + WS_IDX);
  float*    parts  = (float*)   (ws + WS_PART);

  float* q_out   = out;
  float* idx_f   = out + QE;
  float* loss_pt = out + QE + NN;

  // 1. transpose+convert x -> f16 row-major (B*T, C)
  vq_transpose_x<<<dim3((TT + 31) / 32, CC / 32, BB), dim3(32, 32), 0, stream>>>(x, xt);
  // 2. codebook f16 + squared norms
  vq_cb_prep<<<KK, 256, 0, stream>>>(cb, cbh, enorm);
  // 3. WMMA distance GEMM + fused argmin (8 waves / block, 1 M-tile / wave)
  vq_wmma_argmin<<<(MT + 7) / 8, 256, 0, stream>>>(xt, cbh, enorm, idx_ws, idx_f);
  // 4. gather quantized output + per-block loss partials
  vq_gather<<<QE / 256, 256, 0, stream>>>(x, cb, idx_ws, q_out, parts);
  // 5. deterministic mean reduction of commitment loss
  vq_loss_final<<<1, 256, 0, stream>>>(parts, loss_pt);
}
